// DeepONet_66116726555034
// MI455X (gfx1250) — matrix operand order repaired and checked by reading, not compiled, not run on hardware
//
#include <hip/hip_runtime.h>
#include <hip/hip_bf16.h>

typedef _Float16 f16;
typedef __attribute__((ext_vector_type(16))) _Float16 v16h;
typedef __attribute__((ext_vector_type(8)))  float    v8f;
typedef __attribute__((ext_vector_type(4)))  unsigned int u32x4;

#define NB1   16      // N
#define MROWS 16384   // M
#define TDIM  100     // T
#define HID   100
#define KP    128     // padded K
#define TP    112     // padded channel dim (7 tiles of 16)
#define MT    (MROWS/16)   // 1024 m-tiles
#define TT    (TP/16)      // 7 t-tiles
#define MTB   (MT/8)       // m-tile blocks (8 waves per block, 1 m-tile each)
#define EPSV  1e-5f

// ---------------------------------------------------------------------------
// Trunk layer 1: points[M][2] @ W1^T[100][2] -> G[M][100]  (bias cancels in BN)
__global__ void k_trunk_l1(const float* __restrict__ pts,
                           const float* __restrict__ W,
                           float* __restrict__ G) {
    int r = blockIdx.x * blockDim.x + threadIdx.x;
    if (r >= MROWS) return;
    float px = pts[2 * r], py = pts[2 * r + 1];
    float* g = G + (size_t)r * HID;
    #pragma unroll 4
    for (int j = 0; j < HID; ++j)
        g[j] = W[2 * j] * px + W[2 * j + 1] * py;
}

// Per-column batch mean / rstd over MROWS rows. blockIdx.x = column j.
__global__ void k_col_stats(const float* __restrict__ G, float* __restrict__ stats) {
    int j = blockIdx.x;
    __shared__ float rs[256], rq[256];
    float s = 0.f, q = 0.f;
    for (int r = threadIdx.x; r < MROWS; r += 256) {
        float v = G[(size_t)r * HID + j];
        s += v; q += v * v;
    }
    rs[threadIdx.x] = s; rq[threadIdx.x] = q;
    __syncthreads();
    for (int k = 128; k > 0; k >>= 1) {
        if (threadIdx.x < k) { rs[threadIdx.x] += rs[threadIdx.x + k]; rq[threadIdx.x] += rq[threadIdx.x + k]; }
        __syncthreads();
    }
    if (threadIdx.x == 0) {
        float m = rs[0] / (float)MROWS;
        float v = rq[0] / (float)MROWS - m * m;
        stats[j] = m;
        stats[HID + j] = rsqrtf(v + EPSV);
    }
}

// BN + ReLU + cast f32[M][100] -> f16[M][128] (zero-padded K)
__global__ void k_bn_relu_cast(const float* __restrict__ G, const float* __restrict__ stats,
                               const float* __restrict__ ga, const float* __restrict__ be,
                               f16* __restrict__ Ah) {
    int idx = blockIdx.x * blockDim.x + threadIdx.x;
    if (idx >= MROWS * KP) return;
    int r = idx >> 7, j = idx & 127;
    float o = 0.f;
    if (j < HID) {
        float v = (G[(size_t)r * HID + j] - stats[j]) * stats[HID + j] * ga[j] + be[j];
        o = fmaxf(v, 0.f);
    }
    Ah[idx] = (f16)o;
}

// Final trunk layer: add bias, cast to padded f16
__global__ void k_cast_bias(const float* __restrict__ G, const float* __restrict__ b,
                            f16* __restrict__ Ah) {
    int idx = blockIdx.x * blockDim.x + threadIdx.x;
    if (idx >= MROWS * KP) return;
    int r = idx >> 7, j = idx & 127;
    Ah[idx] = (f16)((j < HID) ? (G[(size_t)r * HID + j] + b[j]) : 0.f);
}

// Cast weight f32[100][100] -> f16[112][128] zero-padded (row t contiguous in k)
__global__ void k_wcast(const float* __restrict__ W, f16* __restrict__ Wh) {
    int idx = blockIdx.x * blockDim.x + threadIdx.x;
    if (idx >= TP * KP) return;
    int t = idx >> 7, k = idx & 127;
    Wh[idx] = (f16)((t < HID && k < HID) ? W[t * HID + k] : 0.f);
}

// ---------------------------------------------------------------------------
// Batched WMMA GEMM: Out[b][m][t(<100)] = sum_k A[m][k] * B[b][t][k]
// A: f16 [MROWS][KP] (global), B: f16 [nB][TP][KP] staged in LDS per block.
// Grid: nB * MTB blocks of 256 threads; wave w owns m-tile (blk%MTB)*8+w and
// sweeps all 7 t-tiles, reusing its 4 A fragments. All waves fully active ->
// EXEC all-ones around every WMMA. NT=true streams the write-once output.
template <bool NT>
__global__ void k_wmma_gemm(const f16* __restrict__ A, const f16* __restrict__ B,
                            float* __restrict__ Out, long strideB, long strideOut) {
    __shared__ f16 Bs[TP * KP];           // 28 KB
    const int n   = blockIdx.x / MTB;
    const int mtb = blockIdx.x % MTB;

    // cooperative stage of B matrix (C_n or Wh) into LDS: 7 x b128 per thread
    {
        const u32x4* Bg = (const u32x4*)(B + (size_t)n * strideB);
        u32x4* Bl = (u32x4*)Bs;
        #pragma unroll
        for (int i = 0; i < (TP * KP) / (8 * 256); ++i)
            Bl[i * 256 + threadIdx.x] = Bg[i * 256 + threadIdx.x];
    }
    __syncthreads();

    const int wave = threadIdx.x >> 5;
    const int lane = threadIdx.x & 31;
    const int half = lane >> 4;
    const int l    = lane & 15;
    const int mt   = mtb * 8 + wave;

    union Frag { v16h v; u32x4 q[2]; };

    // A 16x32 f16 layout: lanes<16 hold K {k0..k0+7, k0+16..k0+23}; lanes>=16 shifted by 8
    const f16* Ar = A + (size_t)(mt * 16 + l) * KP;
    Frag af[4];
    #pragma unroll
    for (int s = 0; s < 4; ++s) {
        af[s].q[0] = *reinterpret_cast<const u32x4*>(Ar + s * 32 + half * 8);
        af[s].q[1] = *reinterpret_cast<const u32x4*>(Ar + s * 32 + 16 + half * 8);
    }

    float* OutBase = Out + (size_t)n * strideOut + (size_t)(mt * 16 + half * 8) * TDIM + l;

    #pragma unroll
    for (int tt = 0; tt < TT; ++tt) {
        // B 32x16 f16 layout: lanes<16 hold K {k0..k0+7, k0+16..k0+23}; lanes>=16 shifted by 8 (from LDS)
        const f16* Br = Bs + (size_t)(tt * 16 + l) * KP + half * 8;
        v8f acc = {};
        #pragma unroll
        for (int s = 0; s < 4; ++s) {
            Frag bf;
            bf.q[0] = *reinterpret_cast<const u32x4*>(Br + s * 32);
            bf.q[1] = *reinterpret_cast<const u32x4*>(Br + s * 32 + 16);
            acc = __builtin_amdgcn_wmma_f32_16x16x32_f16(false, af[s].v, false, bf.v,
                                                         (short)0, acc, false, false);
        }
        const int tcol = tt * 16 + l;
        if (tcol < TDIM) {                 // tile 6 keeps only 4 of 16 columns
            float* Or = OutBase + tt * 16;
            #pragma unroll
            for (int r = 0; r < 8; ++r) {  // D: VGPR r -> row r + half*8
                if (NT) __builtin_nontemporal_store(acc[r], Or + (size_t)r * TDIM);
                else    Or[(size_t)r * TDIM] = acc[r];
            }
        }
    }
}

// ---------------------------------------------------------------------------
// Branch1 layer 1: h[16][100] = x[16][16384] @ W1^T. One block per out-channel j.
__global__ void k_b1_l1(const float* __restrict__ x, const float* __restrict__ W,
                        float* __restrict__ h) {
    int j = blockIdx.x;
    __shared__ float red[256];
    float acc[NB1];
    #pragma unroll
    for (int n = 0; n < NB1; ++n) acc[n] = 0.f;
    const float* Wr = W + (size_t)j * 16384;
    for (int i = threadIdx.x; i < 16384; i += 256) {
        float w = Wr[i];
        #pragma unroll
        for (int n = 0; n < NB1; ++n) acc[n] += w * x[n * 16384 + i];
    }
    for (int n = 0; n < NB1; ++n) {
        red[threadIdx.x] = acc[n];
        __syncthreads();
        for (int k = 128; k > 0; k >>= 1) {
            if (threadIdx.x < k) red[threadIdx.x] += red[threadIdx.x + k];
            __syncthreads();
        }
        if (threadIdx.x == 0) h[n * HID + j] = red[0];
        __syncthreads();
    }
}

// Branch2 layer 1: h[100][100] = times[100][1] @ W1^T[100][1]
__global__ void k_b2_l1(const float* __restrict__ tms, const float* __restrict__ W,
                        float* __restrict__ h) {
    int idx = blockIdx.x * blockDim.x + threadIdx.x;
    if (idx >= TDIM * HID) return;
    int r = idx / HID, j = idx % HID;
    h[idx] = tms[r] * W[j];
}

// Small-MLP tail for R<=100 rows in one workgroup: 3 x (BN+ReLU -> Linear), bias on last.
__global__ void k_branch_tail(const float* __restrict__ hin, int R,
                              const float* g0, const float* be0, const float* W2,
                              const float* g1, const float* be1, const float* W3,
                              const float* g2, const float* be2, const float* W4,
                              const float* b4, float* __restrict__ out) {
    __shared__ float act[HID * HID];
    __shared__ float ms[HID], rs[HID];
    const int tid = threadIdx.x, NT = 256;
    const int tot = R * HID;
    for (int i = tid; i < tot; i += NT) act[i] = hin[i];
    __syncthreads();
    const float* Gs[3] = {g0, g1, g2};
    const float* Bs[3] = {be0, be1, be2};
    const float* Ws[3] = {W2, W3, W4};
    for (int l = 0; l < 3; ++l) {
        if (tid < HID) {
            float s = 0.f, q = 0.f;
            for (int r = 0; r < R; ++r) { float v = act[r * HID + tid]; s += v; q += v * v; }
            float m = s / (float)R;
            ms[tid] = m;
            rs[tid] = rsqrtf(q / (float)R - m * m + EPSV);
        }
        __syncthreads();
        for (int i = tid; i < tot; i += NT) {
            int j = i % HID;
            float v = (act[i] - ms[j]) * rs[j] * Gs[l][j] + Bs[l][j];
            act[i] = fmaxf(v, 0.f);
        }
        __syncthreads();
        float vals[40]; int c = 0;
        for (int i = tid; i < tot; i += NT) {
            int r = i / HID, jj = i % HID;
            float a = (l == 2) ? b4[jj] : 0.f;
            const float* Wr = Ws[l] + jj * HID;
            for (int k = 0; k < HID; ++k) a += act[r * HID + k] * Wr[k];
            vals[c++] = a;
        }
        __syncthreads();
        c = 0;
        for (int i = tid; i < tot; i += NT) act[i] = vals[c++];
        __syncthreads();
    }
    for (int i = tid; i < tot; i += NT) out[i] = act[i];
}

// C[n][t][k] = b1[n][k] * b2[t][k], f16, zero-padded to [16][112][128]
__global__ void k_make_C(const float* __restrict__ b1o, const float* __restrict__ b2o,
                         f16* __restrict__ C) {
    int idx = blockIdx.x * blockDim.x + threadIdx.x;
    if (idx >= NB1 * TP * KP) return;
    int n = idx / (TP * KP);
    int rem = idx % (TP * KP);
    int t = rem >> 7, k = rem & 127;
    float v = (t < TDIM && k < HID) ? b1o[n * HID + k] * b2o[t * HID + k] : 0.f;
    C[idx] = (f16)v;
}

// ---------------------------------------------------------------------------
extern "C" void kernel_launch(void* const* d_in, const int* in_sizes, int n_in,
                              void* d_out, int out_size, void* d_ws, size_t ws_size,
                              hipStream_t stream) {
    (void)in_sizes; (void)n_in; (void)out_size; (void)ws_size;
    #define P(i) ((const float*)d_in[i])
    const float* x      = P(0);
    const float* points = P(1);
    const float* times  = P(2);
    // branch1: 3..16, branch2: 17..30, trunk: 31..44 (W,b,g,be per BN'd layer; b1..b3 cancel)

    char* ws = (char*)d_ws;
    size_t off = 0;
    auto alloc = [&](size_t bytes) { void* p = ws + off; off += (bytes + 255) & ~size_t(255); return p; };
    f16*   A0    = (f16*)  alloc((size_t)MROWS * KP * 2);
    f16*   A1    = (f16*)  alloc((size_t)MROWS * KP * 2);
    float* G     = (float*)alloc((size_t)MROWS * HID * 4);
    f16*   Wh    = (f16*)  alloc((size_t)TP * KP * 2);
    float* stats = (float*)alloc(2 * HID * 4);
    float* B1o   = (float*)alloc(NB1 * HID * 4);
    float* B2o   = (float*)alloc(TDIM * HID * 4);
    f16*   Cn    = (f16*)  alloc((size_t)NB1 * TP * KP * 2);
    float* S1    = (float*)alloc(NB1 * HID * 4);
    float* S2    = (float*)alloc(TDIM * HID * 4);

    const int castBlocks = (MROWS * KP) / 256;      // 8192

    // ---- trunk MLP ----
    k_trunk_l1<<<MROWS / 256, 256, 0, stream>>>(points, P(31), G);
    k_col_stats<<<HID, 256, 0, stream>>>(G, stats);
    k_bn_relu_cast<<<castBlocks, 256, 0, stream>>>(G, stats, P(33), P(34), A0);
    k_wcast<<<(TP * KP) / 256, 256, 0, stream>>>(P(35), Wh);
    k_wmma_gemm<false><<<MTB, 256, 0, stream>>>(A0, Wh, G, 0, 0);
    k_col_stats<<<HID, 256, 0, stream>>>(G, stats);
    k_bn_relu_cast<<<castBlocks, 256, 0, stream>>>(G, stats, P(37), P(38), A1);
    k_wcast<<<(TP * KP) / 256, 256, 0, stream>>>(P(39), Wh);
    k_wmma_gemm<false><<<MTB, 256, 0, stream>>>(A1, Wh, G, 0, 0);
    k_col_stats<<<HID, 256, 0, stream>>>(G, stats);
    k_bn_relu_cast<<<castBlocks, 256, 0, stream>>>(G, stats, P(41), P(42), A0);
    k_wcast<<<(TP * KP) / 256, 256, 0, stream>>>(P(43), Wh);
    k_wmma_gemm<false><<<MTB, 256, 0, stream>>>(A0, Wh, G, 0, 0);
    k_cast_bias<<<castBlocks, 256, 0, stream>>>(G, P(44), A1);    // tr in f16 -> A1

    // ---- branch MLPs ----
    k_b1_l1<<<HID, 256, 0, stream>>>(x, P(3), S1);
    k_branch_tail<<<1, 256, 0, stream>>>(S1, NB1,
        P(5), P(6), P(7), P(9), P(10), P(11), P(13), P(14), P(15), P(16), B1o);
    k_b2_l1<<<(TDIM * HID + 255) / 256, 256, 0, stream>>>(times, P(17), S2);
    k_branch_tail<<<1, 256, 0, stream>>>(S2, TDIM,
        P(19), P(20), P(21), P(23), P(24), P(25), P(27), P(28), P(29), P(30), B2o);

    // ---- fused contraction: out[n] = tr @ C_n ----
    k_make_C<<<(NB1 * TP * KP) / 256, 256, 0, stream>>>(B1o, B2o, Cn);
    k_wmma_gemm<true><<<NB1 * MTB, 256, 0, stream>>>(A1, Cn, (float*)d_out,
                                                     (long)TP * KP, (long)MROWS * TDIM);
    #undef P
}
